// VectorQuantizer_16956530885036
// MI455X (gfx1250) — compile-verified, matrix operand order checked by reading, not run on hardware
//
#include <hip/hip_runtime.h>
#include <hip/hip_bf16.h>
#include <math.h>

typedef __attribute__((ext_vector_type(2))) float v2f;
typedef __attribute__((ext_vector_type(8))) float v8f;

#define NUM_EMB   1024
#define EMB_DIM   512
#define N_TOK     32768              // 32 * 32 * 32
#define ZQ_ELEMS  16777216           // 32 * 512 * 32 * 32
#define LOSS_OFF  (ZQ_ELEMS)
#define PERP_OFF  (ZQ_ELEMS + 1)
#define ENC_OFF   (ZQ_ELEMS + 2)

// ws layout (floats): [0] = loss accumulator, [1..1024] = counts (as uint),
//                     [1025..2048] = ||embedding_k||^2

#if __has_builtin(__builtin_amdgcn_wmma_f32_16x16x4_f32)
#define USE_F32_WMMA 1
#else
typedef __attribute__((ext_vector_type(16))) __bf16 v16bf;
#endif

// ---------------------------------------------------------------------------
// Kernel 1: per-code squared norms + zero accumulators.
// ---------------------------------------------------------------------------
__global__ __launch_bounds__(128) void vq_init_kernel(const float* __restrict__ emb,
                                                      float* __restrict__ ws) {
    const int k = blockIdx.x;          // 0..1023
    const int tid = threadIdx.x;       // 128 threads
    const float* row = emb + (long)k * EMB_DIM;
    float s = 0.f;
    for (int d = tid; d < EMB_DIM; d += 128) { float v = row[d]; s += v * v; }
    for (int off = 16; off; off >>= 1) s += __shfl_down(s, off, 32);
    __shared__ float part[4];
    if ((tid & 31) == 0) part[tid >> 5] = s;
    __syncthreads();
    if (tid == 0) {
        ws[1025 + k] = part[0] + part[1] + part[2] + part[3];
        ((unsigned int*)ws)[1 + k] = 0u;   // histogram count
        if (k == 0) ws[0] = 0.f;           // loss accumulator
    }
}

// ---------------------------------------------------------------------------
// Kernel 2: WMMA distance GEMM + argmin + all large outputs.
// One workgroup (256 thr = 8 waves) per 16-token tile.
// ---------------------------------------------------------------------------
__global__ __launch_bounds__(256) void vq_main_kernel(const float* __restrict__ ze,
                                                      const float* __restrict__ emb,
                                                      float* __restrict__ out,
                                                      float* ws) {
    __shared__ float lds_a[16 * 516];    // 16 rows x 512 d, stride 516 (bank-conflict free)
    __shared__ float s_rowNorm[16];
    __shared__ float s_wscore[8][16];
    __shared__ int   s_wcode[8][16];
    __shared__ int   s_idx[16];

    const int tid  = threadIdx.x;
    const int wave = tid >> 5;
    const int lane = tid & 31;
    const int col  = lane & 15;          // WMMA column (code within tile) / A row
    const int hf   = lane >> 4;          // half-wave selector

    const int n_base = blockIdx.x << 4;  // 16 tokens per block
    const int b  = n_base >> 10;
    const int h  = (n_base >> 5) & 31;
    const int w0 = n_base & 31;          // rows are 16 consecutive w positions
    const long zbase = ((long)b * EMB_DIM) * 1024 + h * 32 + w0; // + d*1024 + row

    if (tid < 16) s_rowNorm[tid] = 0.f;
    __syncthreads();

    // ---- Phase 1: stage A tile [16 rows x 512 d] into LDS, accumulate ||x||^2
    {
        const int row = tid & 15;        // constant per thread (256 % 16 == 0)
        float acc = 0.f;
        for (int i = tid; i < 16 * EMB_DIM; i += 256) {
            const int d = i >> 4;
            const float v = ze[zbase + (long)d * 1024 + row]; // 16 consecutive w: coalesced
            lds_a[row * 516 + d] = v;
            acc += v * v;
        }
        atomicAdd(&s_rowNorm[row], acc);
    }
    __syncthreads();

    // ---- Phase 2: distances via WMMA; per-lane running argmin over 8 k-tiles
    float best_s[8];
    int   best_k[8];
#pragma unroll
    for (int v = 0; v < 8; ++v) { best_s[v] = 3.4e38f; best_k[v] = 0; }

    const float* aBase = &lds_a[col * 516 + 2 * hf];

    for (int kt = 0; kt < 8; ++kt) {
        const int kb   = (((kt << 3) + wave) << 4);
        const int code = kb + col;
        const float* bBase = emb + (long)code * EMB_DIM + 2 * hf;

        v8f c = {0.f, 0.f, 0.f, 0.f, 0.f, 0.f, 0.f, 0.f};
#ifdef USE_F32_WMMA
#pragma unroll 4
        for (int s = 0; s < 128; ++s) {   // K-chunk = 4 per wmma
            v2f a  = *(const v2f*)(aBase + 4 * s);   // lanes<16: K=0,1  lanes>=16: K=2,3
            v2f bb = *(const v2f*)(bBase + 4 * s);
            c = __builtin_amdgcn_wmma_f32_16x16x4_f32(
                    false, a, false, bb, (short)0, c, false, false);
        }
#else
        const float* aRow = &lds_a[col * 516];
        const float* bRow = emb + (long)code * EMB_DIM;
        for (int s = 0; s < 16; ++s) {    // K-chunk = 32 per wmma (bf16 fallback)
            const int dbase = s << 5;
            v16bf a, bv;
#pragma unroll
            for (int j = 0; j < 16; ++j) {
                const int koff = j + (hf << 3) + (j & 8);
                a[j]  = (__bf16)aRow[dbase + koff];
                bv[j] = (__bf16)bRow[dbase + koff];
            }
            c = __builtin_amdgcn_wmma_f32_16x16x32_bf16(
                    false, a, false, bv, (short)0, c, false, false);
        }
#endif
        const float en = ws[1025 + code];       // ||e_code||^2
#pragma unroll
        for (int v = 0; v < 8; ++v) {
            const float sc = en - 2.f * c[v];   // k-dependent part of distance
            if (sc < best_s[v] || (sc == best_s[v] && code < best_k[v])) {
                best_s[v] = sc; best_k[v] = code;
            }
        }
    }

    // ---- 16-lane argmin reduction (each half-wave covers one set of 8 rows)
#pragma unroll
    for (int v = 0; v < 8; ++v) {
        float s = best_s[v];
        int   k = best_k[v];
        for (int off = 8; off; off >>= 1) {
            const float os = __shfl_xor(s, off, 16);
            const int   ok = __shfl_xor(k, off, 16);
            if (os < s || (os == s && ok < k)) { s = os; k = ok; }
        }
        if (col == 0) {
            const int M = v + (hf << 3);        // row index 0..15
            s_wscore[wave][M] = s;
            s_wcode[wave][M]  = k;
        }
    }
    __syncthreads();

    // ---- Cross-wave argmin; loss + histogram contributions
    if (tid < 16) {
        float s = s_wscore[0][tid];
        int   k = s_wcode[0][tid];
        for (int wv = 1; wv < 8; ++wv) {
            const float os = s_wscore[wv][tid];
            const int   ok = s_wcode[wv][tid];
            if (os < s || (os == s && ok < k)) { s = os; k = ok; }
        }
        s_idx[tid] = k;
        atomicAdd(&ws[0], s + s_rowNorm[tid]);          // min distance = sum((zq-ze)^2)
        atomicAdd((unsigned int*)ws + 1 + k, 1u);       // code usage count
    }
    __syncthreads();

    // ---- Phase 3a: z_q gather back to [B, D, H, W] (16-wide contiguous w writes)
    {
        const int w_off = tid & 15;
        const int dd    = tid >> 4;       // 0..15
        const int code  = s_idx[w_off];
        for (int db = 0; db < 32; ++db) {
            const int d = (db << 4) + dd;
            out[zbase + (long)d * 1024 + w_off] = emb[(long)code * EMB_DIM + d];
        }
    }

    // ---- Phase 3b: one-hot encodings rows [16 x 1024]
    {
        float* enc = out + ENC_OFF + (long)n_base * NUM_EMB;  // 8-byte aligned
        const v2f z2 = {0.f, 0.f};
        for (int i = tid; i < (16 * NUM_EMB) / 2; i += 256)
            ((v2f*)enc)[i] = z2;
        __syncthreads();
        if (tid < 16) enc[tid * NUM_EMB + s_idx[tid]] = 1.0f;
    }
}

// ---------------------------------------------------------------------------
// Kernel 3: perplexity + loss scalars.
// ---------------------------------------------------------------------------
__global__ __launch_bounds__(256) void vq_final_kernel(const float* __restrict__ ws,
                                                       float* __restrict__ out) {
    __shared__ float part[8];
    const int tid = threadIdx.x;        // 256 threads
    float s = 0.f;
    for (int k = tid; k < NUM_EMB; k += 256) {
        const float p = (float)((const unsigned int*)ws)[1 + k] * (1.0f / (float)N_TOK);
        s += p * logf(p + 1e-10f);
    }
    for (int off = 16; off; off >>= 1) s += __shfl_down(s, off, 32);
    if ((tid & 31) == 0) part[tid >> 5] = s;
    __syncthreads();
    if (tid == 0) {
        float t = 0.f;
        for (int i = 0; i < 8; ++i) t += part[i];
        out[PERP_OFF] = expf(-t);
        out[LOSS_OFF] = 0.25f * ws[0] / (float)((long)N_TOK * EMB_DIM);
    }
}

// ---------------------------------------------------------------------------
extern "C" void kernel_launch(void* const* d_in, const int* in_sizes, int n_in,
                              void* d_out, int out_size, void* d_ws, size_t ws_size,
                              hipStream_t stream) {
    const float* ze  = (const float*)d_in[0];   // [32, 512, 32, 32] fp32
    const float* emb = (const float*)d_in[1];   // [1024, 512] fp32
    float* out = (float*)d_out;
    float* ws  = (float*)d_ws;                  // needs ~8.2 KB

    vq_init_kernel <<<NUM_EMB,     128, 0, stream>>>(emb, ws);
    vq_main_kernel <<<N_TOK / 16,  256, 0, stream>>>(ze, emb, out, ws);
    vq_final_kernel<<<1,           256, 0, stream>>>(ws, out);
}
